// MultiHeadAttention_3667902071328
// MI455X (gfx1250) — matrix operand order repaired and checked by reading, not compiled, not run on hardware
//
#include <hip/hip_runtime.h>
#include <hip/hip_bf16.h>

// ---------------------------------------------------------------------------
// MHA forward for MI455X (gfx1250, wave32, WMMA).
// B=4, SQ=SK=2048, EMB=1024, HEAD=16, QK=V=64.
// Pipeline: weight transpose->bf16, 3x projection GEMM (bf16 WMMA, f32 acc),
// flash-attention (QK^T, online softmax, P*V) all in bf16 WMMA.
// Data movement: async global->LDS (GLOBAL_LOAD_ASYNC_TO_LDS_B128),
// global_prefetch for next tiles, padded LDS to kill bank conflicts.
// ---------------------------------------------------------------------------

#define Bc    4
#define SQc   2048
#define SKc   2048
#define EMBc  1024
#define Hc    16
#define Dc    64
#define NPROJ 1024   // HEAD*QK = HEAD*V

typedef __attribute__((ext_vector_type(16))) __bf16 bf16x16;
typedef __attribute__((ext_vector_type(8)))  float  f32x8;
typedef __attribute__((ext_vector_type(4)))  int    v4i;

union FragU {
    uint4   u4[2];
    bf16x16 v;
};

// ---------------------------------------------------------------------------
// Async global->LDS copy (GLOBAL_LOAD_ASYNC_TO_LDS_B128, ASYNCcnt-tracked).
// Probe-derived signature: (v4i AS(1)*, v4i AS(3)*, imm offset, imm cpol).
// Guarded: falls back to load+ds_store if the builtin is not declared.
// ---------------------------------------------------------------------------
#if defined(__has_builtin) && __has_builtin(__builtin_amdgcn_global_load_async_to_lds_b128)
#define HAS_ASYNC_LDS 1
#else
#define HAS_ASYNC_LDS 0
#endif

#if HAS_ASYNC_LDS
typedef __attribute__((address_space(1))) v4i as1_v4i;
typedef __attribute__((address_space(3))) v4i as3_v4i;
#endif

static __device__ __forceinline__ void copy16_g2lds(const void* g, void* l) {
#if HAS_ASYNC_LDS
    // Generic->AS casts via integer: global generic bits == AS(1) address;
    // LDS offset == low 32 bits of the generic (aperture-based) pointer.
    as1_v4i* gp = (as1_v4i*)(unsigned long long)g;
    as3_v4i* lp = (as3_v4i*)(unsigned int)(unsigned long long)l;
    __builtin_amdgcn_global_load_async_to_lds_b128(gp, lp, 0, 0);
#else
    *(uint4*)l = *(const uint4*)g;
#endif
}

static __device__ __forceinline__ void async_wait0() {
#if HAS_ASYNC_LDS
#if defined(__has_builtin) && __has_builtin(__builtin_amdgcn_s_wait_asynccnt)
    __builtin_amdgcn_s_wait_asynccnt(0);
#else
    asm volatile("s_wait_asynccnt 0" ::: "memory");
#endif
#endif
}

static __device__ __forceinline__ unsigned short f2bf(float f) {
    unsigned int u = __float_as_uint(f);
    unsigned int r = u + 0x7FFFu + ((u >> 16) & 1u);   // round-to-nearest-even
    return (unsigned short)(r >> 16);
}

static __device__ __forceinline__ f32x8 wmma_bf16(bf16x16 a, bf16x16 b, f32x8 c) {
    // D = A(16x32 bf16) * B(32x16 bf16) + C(16x16 f32)
    return __builtin_amdgcn_wmma_f32_16x16x32_bf16(
        /*neg_a=*/false, a, /*neg_b=*/false, b,
        /*c_mod=*/(short)0, c, /*reuse_a=*/false, /*reuse_b=*/false);
}

// A-matrix fragment, 16x32 bf16 (MxK). Lane holds row M=lane%16.
// VGPR0-3: K = 8*half + [0..7]; VGPR4-7: K = 16 + 8*half + [0..7].
static __device__ __forceinline__ bf16x16
load_a_frag(const unsigned short* base, int stride, int row, int half) {
    FragU f;
    const unsigned short* p = base + row * stride;
    f.u4[0] = *(const uint4*)(p + 8 * half);
    f.u4[1] = *(const uint4*)(p + 16 + 8 * half);
    return f.v;
}

// B-matrix fragment, 32x16 bf16 (KxN), sourced from an N-major [N][K] tile.
// Lane holds column N=lane%16; VGPR0-3: K = 8*half + [0..7]; VGPR4-7: K = 16 + 8*half + [0..7].
static __device__ __forceinline__ bf16x16
load_b_frag(const unsigned short* base, int stride, int col, int half) {
    FragU f;
    const unsigned short* p = base + col * stride;
    f.u4[0] = *(const uint4*)(p + 8 * half);
    f.u4[1] = *(const uint4*)(p + 16 + 8 * half);
    return f.v;
}

// ---------------------------------------------------------------------------
// Kernel 1: W [K=1024][N=1024] f32  ->  W^T [N][K] bf16 (LDS tiled transpose)
// ---------------------------------------------------------------------------
__global__ void __launch_bounds__(256)
transpose_cast_kernel(const float* __restrict__ in, unsigned short* __restrict__ out) {
    __shared__ float tile[32][33];
    const int kb = blockIdx.x * 32;
    const int nb = blockIdx.y * 32;
    const int tx = threadIdx.x & 31;
    const int ty = threadIdx.x >> 5;          // 0..7
#pragma unroll
    for (int i = ty; i < 32; i += 8)
        tile[i][tx] = in[(size_t)(kb + i) * NPROJ + nb + tx];
    __syncthreads();
#pragma unroll
    for (int i = ty; i < 32; i += 8)
        out[(size_t)(nb + i) * EMBc + kb + tx] = f2bf(tile[tx][i]);
}

// ---------------------------------------------------------------------------
// Kernel 2: projection GEMM  C[M=8192, N=1024] = X f32 @ W + bias
// Block: 256 threads (8 waves), tile M=128 x N=64, K step 32.
// Wave w computes rows [16w,16w+16) x all 64 columns (4 WMMA n-tiles).
// mode 0/1: dst[((b*H+h)*S+s)*64+d]   (Q,K row-major per head)
// mode 2  : dst[((b*H+h)*64+d)*SK+s]  (V transposed: [d][key])
// ---------------------------------------------------------------------------
__global__ void __launch_bounds__(256)
proj_gemm_kernel(const float* __restrict__ X, const unsigned short* __restrict__ Wt,
                 const float* __restrict__ bias, unsigned short* __restrict__ dst,
                 int mode) {
    __shared__ unsigned short Xs[128 * 40];   // 128 rows x 32 K, stride 40 (pad)
    __shared__ unsigned short Ws[64 * 40];    // 64 cols  x 32 K, stride 40 (pad)

    const int tid  = threadIdx.x;
    const int wave = tid >> 5;
    const int lane = tid & 31;
    const int half = lane >> 4;
    const int ln   = lane & 15;
    const int mbase = blockIdx.y * 128;
    const int nbase = blockIdx.x * 64;

    f32x8 acc[4];
#pragma unroll
    for (int t = 0; t < 4; ++t) acc[t] = 0.0f;

    for (int kb = 0; kb < EMBc; kb += 32) {
        // Stage W^T tile via async global->LDS: 64x32 bf16 = 256 x 16B chunks.
        {
            int r  = tid >> 2;          // 0..63
            int c4 = tid & 3;           // 0..3 (8 ushorts each)
            copy16_g2lds(Wt + (size_t)(nbase + r) * EMBc + kb + c4 * 8,
                         Ws + r * 40 + c4 * 8);
        }
        // Prefetch next W^T K-slice into L2 while we convert X.
        if (kb + 32 < EMBc)
            __builtin_prefetch(Wt + (size_t)(nbase + (tid >> 2)) * EMBc + kb + 32, 0, 1);

        // Stage X tile: cast f32 -> bf16 on the fly. 128x32 f32 = 1024 float4.
#pragma unroll
        for (int i = 0; i < 4; ++i) {
            int idx = tid + i * 256;
            int r  = idx >> 3;          // 0..127
            int c4 = idx & 7;           // 0..7 (4 floats each)
            float4 f = *(const float4*)(X + (size_t)(mbase + r) * EMBc + kb + c4 * 4);
            ushort4 h;
            h.x = f2bf(f.x); h.y = f2bf(f.y); h.z = f2bf(f.z); h.w = f2bf(f.w);
            *(ushort4*)(Xs + r * 40 + c4 * 4) = h;
        }
        async_wait0();
        __syncthreads();

        bf16x16 a = load_a_frag(Xs + wave * 16 * 40, 40, ln, half);
#pragma unroll
        for (int nt = 0; nt < 4; ++nt) {
            bf16x16 b = load_b_frag(Ws + nt * 16 * 40, 40, ln, half);
            acc[nt] = wmma_bf16(a, b, acc[nt]);
        }
        __syncthreads();
    }

    // Epilogue: bias add, cast bf16, scatter into attention layout.
#pragma unroll
    for (int nt = 0; nt < 4; ++nt) {
        const int c  = nbase + nt * 16 + ln;     // output column
        const float bv = bias[c];
        const int hh = c >> 6;                   // head
        const int d  = c & 63;                   // dim within head
#pragma unroll
        for (int j = 0; j < 8; ++j) {
            const int r = mbase + wave * 16 + 8 * half + j;   // token row
            const int b = r >> 11;               // r / 2048
            const int s = r & 2047;
            const float v = acc[nt][j] + bv;
            size_t off;
            if (mode == 2)
                off = ((size_t)(b * Hc + hh) * Dc + d) * (size_t)SKc + s;
            else
                off = ((size_t)(b * Hc + hh) * (size_t)SQc + s) * Dc + d;
            dst[off] = f2bf(v);
        }
    }
}

// ---------------------------------------------------------------------------
// Kernel 3: flash attention.
// Grid: (SQ/64, B*H), 128 threads (4 waves). Wave owns 16 q rows.
// K tile (32 keys x 64 d) and V tile (64 d x 32 keys) staged once per block
// into padded LDS (stride 72 -> conflict-free b128 fragment reads).
// Per 32-key chunk: S = Q K^T (2 n-tiles x 2 k-steps), mask -> online softmax
// -> P (bf16, transposed via LDS) -> O += P V (4 d-tiles).
// ---------------------------------------------------------------------------
__global__ void __launch_bounds__(128)
flash_attn_kernel(const unsigned short* __restrict__ Qb,
                  const unsigned short* __restrict__ Kb,
                  const unsigned short* __restrict__ Vt,
                  const unsigned char* __restrict__ mask,
                  float* __restrict__ out) {
    const int qblk = blockIdx.x;
    const int bh   = blockIdx.y;
    const int b    = bh / Hc;
    const int h    = bh % Hc;
    const int tid  = threadIdx.x;
    const int wave = tid >> 5;
    const int lane = tid & 31;
    const int half = lane >> 4;
    const int ln   = lane & 15;
    const int qbase = qblk * 64 + wave * 16;

    __shared__ unsigned short Ks[32 * 72];      // [key][d], stride 72
    __shared__ unsigned short Vs[64 * 72];      // [d][key], stride 72
    __shared__ unsigned short Pst[4][16][48];   // per-wave P tile, padded

    // Resident Q fragments: 16 q rows x 64 d = 2 k-steps of 32.
    const unsigned short* Qrow = Qb + ((size_t)bh * SQc + qbase) * Dc;
    bf16x16 qfrag[2];
#pragma unroll
    for (int ks = 0; ks < 2; ++ks)
        qfrag[ks] = load_a_frag(Qrow + ks * 32, Dc, ln, half);

    f32x8 O[4];
#pragma unroll
    for (int t = 0; t < 4; ++t) O[t] = 0.0f;

    float m_run[8], l_run[8];
#pragma unroll
    for (int j = 0; j < 8; ++j) { m_run[j] = -3.0e38f; l_run[j] = 0.0f; }

    const float scale = 0.125f;   // 1/sqrt(64)
    const unsigned char* mrow = mask + (size_t)b * SQc * SKc;
    const unsigned short* Kbh = Kb + (size_t)bh * SKc * Dc;
    const unsigned short* Vbh = Vt + (size_t)bh * Dc * SKc;

    for (int kc = 0; kc < SKc; kc += 32) {
        // ---- stage K tile: 32 rows x 64 d = 256 x 16B chunks, 2/thread ----
#pragma unroll
        for (int i = 0; i < 2; ++i) {
            int c   = tid + i * 128;
            int row = c >> 3;
            int off = (c & 7) * 8;
            copy16_g2lds(Kbh + (size_t)(kc + row) * Dc + off, Ks + row * 72 + off);
        }
        // ---- stage V tile: 64 rows x 32 keys = 256 x 16B chunks, 2/thread ----
#pragma unroll
        for (int i = 0; i < 2; ++i) {
            int c   = tid + i * 128;
            int row = c >> 2;
            int off = (c & 3) * 8;
            copy16_g2lds(Vbh + (size_t)row * SKc + kc + off, Vs + row * 72 + off);
        }
        // Prefetch next chunk's K/V lines into L2.
        if (kc + 32 < SKc) {
            __builtin_prefetch(Kbh + (size_t)(kc + 32 + (tid >> 2)) * Dc, 0, 1);
            __builtin_prefetch(Vbh + (size_t)(tid) * SKc + kc + 32, 0, 1);
        }
        async_wait0();
        __syncthreads();

        // ---- S = Q K^T : two 16x16 n-tiles from LDS K tile ----
        f32x8 S[2];
#pragma unroll
        for (int nt = 0; nt < 2; ++nt) {
            f32x8 a = 0.0f;
#pragma unroll
            for (int ks = 0; ks < 2; ++ks) {
                bf16x16 kf = load_b_frag(Ks + (nt * 16) * 72 + ks * 32, 72, ln, half);
                a = wmma_bf16(qfrag[ks], kf, a);
            }
            S[nt] = a;
        }

        // ---- scale + mask (mask==true -> -1e9, per reference) ----
#pragma unroll
        for (int nt = 0; nt < 2; ++nt) {
            const int key = kc + nt * 16 + ln;
#pragma unroll
            for (int j = 0; j < 8; ++j) {
                const int q = qbase + 8 * half + j;
                float s = S[nt][j] * scale;
                if (mrow[(size_t)q * SKc + key]) s = -1.0e9f;
                S[nt][j] = s;
            }
        }

        // ---- online softmax: row max/sum across the 16 lanes of a row ----
        float corr[8], rs[8];
#pragma unroll
        for (int j = 0; j < 8; ++j) {
            float v = fmaxf(S[0][j], S[1][j]);
            v = fmaxf(v, __shfl_xor(v, 1, 32));
            v = fmaxf(v, __shfl_xor(v, 2, 32));
            v = fmaxf(v, __shfl_xor(v, 4, 32));
            v = fmaxf(v, __shfl_xor(v, 8, 32));
            const float nm = fmaxf(m_run[j], v);
            corr[j] = __expf(m_run[j] - nm);
            m_run[j] = nm;
            rs[j] = 0.0f;
        }

        // ---- P = exp(S - m); stash bf16 P (transposed) into LDS ----
#pragma unroll
        for (int nt = 0; nt < 2; ++nt) {
#pragma unroll
            for (int j = 0; j < 8; ++j) {
                const float p = __expf(S[nt][j] - m_run[j]);
                rs[j] += p;
                Pst[wave][8 * half + j][nt * 16 + ln] = f2bf(p);
            }
        }
#pragma unroll
        for (int j = 0; j < 8; ++j) {
            float v = rs[j];
            v += __shfl_xor(v, 1, 32);
            v += __shfl_xor(v, 2, 32);
            v += __shfl_xor(v, 4, 32);
            v += __shfl_xor(v, 8, 32);
            l_run[j] = l_run[j] * corr[j] + v;
        }
#pragma unroll
        for (int t = 0; t < 4; ++t)
#pragma unroll
            for (int j = 0; j < 8; ++j) O[t][j] *= corr[j];

        __syncthreads();   // Pst stores visible; Ks reads done

        bf16x16 pfrag = load_a_frag(&Pst[wave][0][0], 48, ln, half);
#pragma unroll
        for (int t = 0; t < 4; ++t) {
            bf16x16 vf = load_b_frag(Vs + (t * 16) * 72, 72, ln, half);
            O[t] = wmma_bf16(pfrag, vf, O[t]);
        }
        __syncthreads();   // Vs/Pst reads done before next-iter overwrite
    }

    // ---- normalize and write [B, SQ, H*V] f32 ----
#pragma unroll
    for (int t = 0; t < 4; ++t) {
        const int d = t * 16 + ln;
#pragma unroll
        for (int j = 0; j < 8; ++j) {
            const int q = qbase + 8 * half + j;
            out[((size_t)b * SQc + q) * (Hc * Dc) + h * Dc + d] = O[t][j] / l_run[j];
        }
    }
}

// ---------------------------------------------------------------------------
// Host launch
// ---------------------------------------------------------------------------
extern "C" void kernel_launch(void* const* d_in, const int* in_sizes, int n_in,
                              void* d_out, int out_size, void* d_ws, size_t ws_size,
                              hipStream_t stream) {
    const float* x_q   = (const float*)d_in[0];
    const float* x_kv  = (const float*)d_in[1];
    const unsigned char* attn_mask = (const unsigned char*)d_in[2];
    const float* w_q = (const float*)d_in[3];
    const float* b_q = (const float*)d_in[4];
    const float* w_k = (const float*)d_in[5];
    const float* b_k = (const float*)d_in[6];
    const float* w_v = (const float*)d_in[7];
    const float* b_v = (const float*)d_in[8];
    float* out = (float*)d_out;

    // Workspace layout (bytes)
    char* ws = (char*)d_ws;
    const size_t WT_BYTES  = (size_t)EMBc * NPROJ * sizeof(unsigned short);          // 2 MB
    const size_t QKV_BYTES = (size_t)Bc * Hc * SQc * Dc * sizeof(unsigned short);    // 16 MB
    unsigned short* WtQ = (unsigned short*)(ws);
    unsigned short* WtK = (unsigned short*)(ws + WT_BYTES);
    unsigned short* WtV = (unsigned short*)(ws + 2 * WT_BYTES);
    unsigned short* Qb  = (unsigned short*)(ws + 3 * WT_BYTES);
    unsigned short* Kb  = (unsigned short*)(ws + 3 * WT_BYTES + QKV_BYTES);
    unsigned short* Vt  = (unsigned short*)(ws + 3 * WT_BYTES + 2 * QKV_BYTES);

    // 1) weights -> bf16 transposed [N][K]
    dim3 tgrid(EMBc / 32, NPROJ / 32);
    transpose_cast_kernel<<<tgrid, 256, 0, stream>>>(w_q, WtQ);
    transpose_cast_kernel<<<tgrid, 256, 0, stream>>>(w_k, WtK);
    transpose_cast_kernel<<<tgrid, 256, 0, stream>>>(w_v, WtV);

    // 2) projections (M = B*S = 8192 rows, N = 1024)
    dim3 pgrid(NPROJ / 64, (Bc * SQc) / 128);
    proj_gemm_kernel<<<pgrid, 256, 0, stream>>>(x_q,  WtQ, b_q, Qb, 0);
    proj_gemm_kernel<<<pgrid, 256, 0, stream>>>(x_kv, WtK, b_k, Kb, 1);
    proj_gemm_kernel<<<pgrid, 256, 0, stream>>>(x_kv, WtV, b_v, Vt, 2);

    // 3) flash attention
    dim3 fgrid(SQc / 64, Bc * Hc);
    flash_attn_kernel<<<fgrid, 128, 0, stream>>>(Qb, Kb, Vt, attn_mask, out);
}
